// GGCN_7129645711852
// MI455X (gfx1250) — compile-verified, matrix operand order checked by reading, not run on hardware
//
#include <hip/hip_runtime.h>
#include <hip/hip_bf16.h>
#include <math.h>

// ---------------------------------------------------------------------------
// GGCN forward for MI455X (gfx1250, wave32, WMMA + async LDS copy).
// Heavy math = v_wmma_f32_16x16x32_bf16 (f32 accumulate).
// Fused attention+propagation kernel avoids materializing any N x N tensor:
// per layer it reads adj (64MB) once (via GLOBAL_LOAD_ASYNC_TO_LDS_B128,
// overlapped with the prod WMMAs) and streams Wh (2MB bf16) from L2.
// ---------------------------------------------------------------------------

typedef __attribute__((ext_vector_type(16))) __bf16 v16bf;
typedef __attribute__((ext_vector_type(8)))  __bf16 v8bf;
typedef __attribute__((ext_vector_type(8)))  float  v8f;

#define NNODE 4096
#define EPSF  1e-9f

// Load one 16-bit WMMA operand fragment (16x32 tile, one matrix row per lane)
// from a row-major bf16 matrix: two contiguous 16-byte chunks.
__device__ __forceinline__ v16bf load_frag16(const __bf16* row, int klo, int khi) {
  v8bf lo = *(const v8bf*)(row + klo);
  v8bf hi = *(const v8bf*)(row + khi);
  return __builtin_shufflevector(lo, hi, 0,1,2,3,4,5,6,7,8,9,10,11,12,13,14,15);
}

__device__ __forceinline__ v8f wmma_bf16(v16bf a, v16bf b, v8f c) {
  return __builtin_amdgcn_wmma_f32_16x16x32_bf16(false, a, false, b, (short)0, c,
                                                 false, false);
}

__device__ __forceinline__ float softplusf(float z) {
  return fmaxf(z, 0.f) + log1pf(expf(-fabsf(z)));
}
__device__ __forceinline__ float eluf(float x) {
  return x > 0.f ? x : expm1f(x);
}

// ---------------------------------------------------------------------------
__global__ void diag_k(const float* __restrict__ adj, float* __restrict__ dg) {
  int i = blockIdx.x * blockDim.x + threadIdx.x;
  if (i < NNODE) dg[i] = adj[(size_t)i * (NNODE + 1)];
}

__global__ void cvt_bf16(const float* __restrict__ src, __bf16* __restrict__ dst, int n) {
  int i = blockIdx.x * blockDim.x + threadIdx.x;
  if (i < n) dst[i] = (__bf16)src[i];
}

// dst[c][r] = bf16(src[r][c])   (src: [R][C] f32)
__global__ void transpose_f32_bf16(const float* __restrict__ src,
                                   __bf16* __restrict__ dst, int R, int C) {
  int idx = blockIdx.x * blockDim.x + threadIdx.x;
  if (idx >= R * C) return;
  int r = idx / C, c = idx - r * C;
  dst[(size_t)c * R + r] = (__bf16)src[idx];
}

// C[M][N] = A[M][K] @ BT[N][K]^T + bias.  One wave per 16x16 output tile.
__global__ void gemm_bias(const __bf16* __restrict__ A, const __bf16* __restrict__ BT,
                          const float* __restrict__ bias, float* __restrict__ C,
                          int M, int Ncols, int K) {
  int wid  = threadIdx.x >> 5;
  int lane = threadIdx.x & 31;
  int t    = blockIdx.x * (blockDim.x >> 5) + wid;
  int ntn  = Ncols >> 4;
  int ti = t / ntn, tj = t - ti * ntn;
  if (ti * 16 >= M) return;
  int ln = lane & 15, half = lane >> 4;
  const __bf16* arow = A  + (size_t)(ti * 16 + ln) * K;
  const __bf16* brow = BT + (size_t)(tj * 16 + ln) * K;
  v8f acc = {0.f,0.f,0.f,0.f,0.f,0.f,0.f,0.f};
  for (int kb = 0; kb < K; kb += 32) {
    v16bf a = load_frag16(arow, kb + half * 8, kb + 16 + half * 8);
    v16bf b = load_frag16(brow, kb + half * 8, kb + 16 + half * 8);
    acc = wmma_bf16(a, b, acc);
  }
  float bb = bias[tj * 16 + ln];
#pragma unroll
  for (int r = 0; r < 8; r++) {
    int row = ti * 16 + r + 8 * half;
    C[(size_t)row * Ncols + tj * 16 + ln] = acc[r] + bb;
  }
}

// Per row: squared norm + zero-padded bf16 copy ([4096][256]).
__global__ void rowpack(const float* __restrict__ wh, int dout,
                        __bf16* __restrict__ whbf, float* __restrict__ sqn) {
  int i = blockIdx.x, j = threadIdx.x;                // 256 threads
  float v = (j < dout) ? wh[(size_t)i * dout + j] : 0.f;
  whbf[(size_t)i * 256 + j] = (__bf16)v;
  __shared__ float red[256];
  red[j] = v * v;
  __syncthreads();
  for (int s = 128; s > 0; s >>= 1) {
    if (j < s) red[j] += red[j + s];
    __syncthreads();
  }
  if (j == 0) sqn[i] = red[0];
}

// ---------------------------------------------------------------------------
// Fused: async-copy adj tile to LDS; prod = Wh@Wh^T tile (WMMA, overlapped
// with the async copy); cosine/degree/softplus attention;
// att_c = c0*relu(att)+c1*min(att,0) (bf16, relayout via LDS);
// prop += att_c @ Wh.  One wave per 16-row tile, 128 j-chunks of 32.
// ---------------------------------------------------------------------------
template <int DOUT>
__global__ void fused_att_prop(const float* __restrict__ adj,
                               const float* __restrict__ dg,
                               const float* __restrict__ sqn,
                               const __bf16* __restrict__ whbf,   // [4096][256] padded
                               const __bf16* __restrict__ whtbf,  // [DOUT][4096]
                               const float* __restrict__ whf,     // [4096][DOUT]
                               float* __restrict__ out,           // [4096][DOUT]
                               const float* __restrict__ dcp,     // 2
                               const float* __restrict__ cfp,     // 3
                               const float* __restrict__ spp) {   // 1
  constexpr int NB  = DOUT / 16;
  constexpr int D32 = (DOUT + 31) / 32;
  int wid  = threadIdx.x >> 5;
  int lane = threadIdx.x & 31;
  int tile = blockIdx.x * 8 + wid;
  int i0   = tile * 16;
  int ln   = lane & 15, half = lane >> 4;

  __shared__ __bf16 s_att[8 * 512];                 // per-wave 16x32 bf16 att tile
  __shared__ __align__(16) float s_adj[8 * 512];    // per-wave 16x32 f32 adj tile
  __bf16* att  = s_att + wid * 512;
  float*  ladj = s_adj + wid * 512;
  // Wave-relative LDS byte offset of this wave's adj tile (low 32 bits of the
  // generic pointer to a __shared__ object are the LDS offset on amdgcn).
  unsigned int lbase = (unsigned int)(size_t)ladj;

  // per-layer scalar params
  float dc0 = dcp[0], dc1 = dcp[1];
  float cm  = fmaxf(cfp[0], fmaxf(cfp[1], cfp[2]));
  float e0 = expf(cfp[0] - cm), e1 = expf(cfp[1] - cm), e2 = expf(cfp[2] - cm);
  float inv = 1.f / (e0 + e1 + e2);
  float c0 = e0 * inv, c1 = e1 * inv, c2 = e2 * inv;
  float s  = softplusf(spp[0]);

  // Cache this tile's A fragments of Wh (constant over the whole j loop).
  v16bf afr[D32];
#pragma unroll
  for (int kb = 0; kb < D32; kb++)
    afr[kb] = load_frag16(whbf + (size_t)(i0 + ln) * 256,
                          kb * 32 + half * 8, kb * 32 + 16 + half * 8);

  float sqi[8], dgi[8];
#pragma unroll
  for (int r = 0; r < 8; r++) {
    int i = i0 + r + 8 * half;
    sqi[r] = sqn[i];
    dgi[r] = dg[i];
  }

  v8f acc[NB];
#pragma unroll
  for (int nb = 0; nb < NB; nb++)
    acc[nb] = (v8f){0.f,0.f,0.f,0.f,0.f,0.f,0.f,0.f};

  for (int j0 = 0; j0 < NNODE; j0 += 32) {
    // --- Issue async copy of this chunk's adj tile (16 rows x 128B) into our
    // wave-private LDS region: 4 x GLOBAL_LOAD_ASYNC_TO_LDS_B128 per lane set.
    // Overlaps with the prod WMMAs below; completion via ASYNCcnt.
#pragma unroll
    for (int q = 0; q < 4; q++) {
      int cc   = lane + q * 32;          // 128 chunks of 16B
      int row  = cc >> 3;                // 8 chunks per 128B row
      int colb = (cc & 7) * 16;          // byte offset within row
      unsigned long long ga = (unsigned long long)(size_t)
          ((const char*)adj + ((size_t)(i0 + row) * NNODE + j0) * 4 + colb);
      unsigned int la = lbase + (unsigned int)(row * 128 + colb);
      asm volatile("global_load_async_to_lds_b128 %0, %1, off"
                   :: "v"(la), "v"(ga) : "memory");
    }

    // --- prod tiles: Wh[i0..][:] @ Wh[j0..j0+31][:]^T  (independent of adj)
    v8f p[2];
#pragma unroll
    for (int t = 0; t < 2; t++) {
      p[t] = (v8f){0.f,0.f,0.f,0.f,0.f,0.f,0.f,0.f};
#pragma unroll
      for (int kb = 0; kb < D32; kb++) {
        v16bf b = load_frag16(whbf + (size_t)(j0 + t * 16 + ln) * 256,
                              kb * 32 + half * 8, kb * 32 + 16 + half * 8);
        p[t] = wmma_bf16(afr[kb], b, p[t]);
      }
    }

    // --- wait for the async adj tile, then elementwise attention math
    asm volatile("s_wait_asynccnt 0" ::: "memory");
#pragma unroll
    for (int t = 0; t < 2; t++) {
      int jj = t * 16;
      int j  = j0 + jj + ln;
      float sqj = sqn[j];
#pragma unroll
      for (int r = 0; r < 8; r++) {
        int i = i0 + r + 8 * half;
        float denom = fmaxf(sqrtf(sqi[r] * sqj), EPSF);
        float e = p[t][r] / denom;
        if (i == j) e = 0.f;
        float av = ladj[(r + 8 * half) * 32 + jj + ln];
        float degp = dgi[r] / fmaxf(av, EPSF) - 1.f;
        float sc = softplusf(dc0 * degp + dc1);
        float a = e * av * sc;
        float ac = c0 * fmaxf(a, 0.f) + c1 * fminf(a, 0.f);
        att[(r + 8 * half) * 32 + jj + ln] = (__bf16)ac;
      }
    }
    __syncthreads();  // uniform across all 8 waves; makes LDS RAW explicit

    // --- propagation: acc += att_c(16x32) @ Wh[j0..j0+31][:]
    v16bf ap = load_frag16(att + ln * 32, half * 8, 16 + half * 8);
#pragma unroll
    for (int nb = 0; nb < NB; nb++) {
      v16bf b = load_frag16(whtbf + (size_t)(nb * 16 + ln) * NNODE,
                            j0 + half * 8, j0 + 16 + half * 8);
      acc[nb] = wmma_bf16(ap, b, acc[nb]);
    }
    __syncthreads();
  }

  // out = s * (prop_combined + c2 * Wh)
#pragma unroll
  for (int nb = 0; nb < NB; nb++) {
#pragma unroll
    for (int r = 0; r < 8; r++) {
      int i = i0 + r + 8 * half;
      int f = nb * 16 + ln;
      out[(size_t)i * DOUT + f] = s * (acc[nb][r] + c2 * whf[(size_t)i * DOUT + f]);
    }
  }
}

// ---------------------------------------------------------------------------
__global__ void elu_k(const float* __restrict__ in, float* __restrict__ out, int n) {
  int i = blockIdx.x * blockDim.x + threadIdx.x;
  if (i < n) out[i] = eluf(in[i]);
}

__global__ void axpy_elu(const float* __restrict__ inner, float* __restrict__ prev,
                         float coeff, int n) {
  int i = blockIdx.x * blockDim.x + threadIdx.x;
  if (i < n) prev[i] = prev[i] + coeff * eluf(inner[i]);
}

__global__ void logsoftmax16(const float* __restrict__ in, float* __restrict__ out) {
  int i = blockIdx.x * blockDim.x + threadIdx.x;
  if (i >= NNODE) return;
  const float* r = in + (size_t)i * 16;
  float mx = r[0];
#pragma unroll
  for (int j = 1; j < 16; j++) mx = fmaxf(mx, r[j]);
  float sum = 0.f;
#pragma unroll
  for (int j = 0; j < 16; j++) sum += expf(r[j] - mx);
  float lse = logf(sum);
#pragma unroll
  for (int j = 0; j < 16; j++) out[(size_t)i * 16 + j] = r[j] - mx - lse;
}

// ---------------------------------------------------------------------------
extern "C" void kernel_launch(void* const* d_in, const int* in_sizes, int n_in,
                              void* d_out, int out_size, void* d_ws, size_t ws_size,
                              hipStream_t stream) {
  (void)in_sizes; (void)n_in; (void)out_size; (void)ws_size;
  const float* x   = (const float*)d_in[0];
  const float* adj = (const float*)d_in[1];
  const float* Wf  = (const float*)d_in[2];
  const float* bff = (const float*)d_in[3];
  const float* W[4] = {(const float*)d_in[4], (const float*)d_in[6],
                       (const float*)d_in[8], (const float*)d_in[10]};
  const float* B[4] = {(const float*)d_in[5], (const float*)d_in[7],
                       (const float*)d_in[9], (const float*)d_in[11]};
  const float* dc = (const float*)d_in[12];   // [4][2]
  const float* cf = (const float*)d_in[13];   // [4][3]
  const float* sp = (const float*)d_in[14];   // [4][1]
  float* out = (float*)d_out;

  char* p = (char*)d_ws;
  auto carve = [&](size_t bytes) -> void* {
    void* q = (void*)p;
    p += (bytes + 255) & ~(size_t)255;
    return q;
  };
  __bf16* xbf   = (__bf16*)carve((size_t)NNODE * 512 * 2);
  __bf16* hbf   = (__bf16*)carve((size_t)NNODE * 256 * 2);
  __bf16* wt    = (__bf16*)carve((size_t)512 * 256 * 2);
  float*  whf   = (float*) carve((size_t)NNODE * 256 * 4);
  __bf16* whbf  = (__bf16*)carve((size_t)NNODE * 256 * 2);
  __bf16* whtbf = (__bf16*)carve((size_t)256 * NNODE * 2);
  float*  sqn   = (float*) carve((size_t)NNODE * 4);
  float*  dg    = (float*) carve((size_t)NNODE * 4);
  float*  prev  = (float*) carve((size_t)NNODE * 256 * 4);
  float*  inner = (float*) carve((size_t)NNODE * 256 * 4);

  diag_k<<<NNODE / 256, 256, 0, stream>>>(adj, dg);
  cvt_bf16<<<(NNODE * 512) / 256, 256, 0, stream>>>(x, xbf, NNODE * 512);

  // feature head: prev = elu(x @ Wf + bf)
  transpose_f32_bf16<<<(512 * 256 + 255) / 256, 256, 0, stream>>>(Wf, wt, 512, 256);
  gemm_bias<<<(NNODE / 16) * (256 / 16) / 8, 256, 0, stream>>>(xbf, wt, bff, whf,
                                                               NNODE, 256, 512);
  elu_k<<<(NNODE * 256) / 256, 256, 0, stream>>>(whf, prev, NNODE * 256);

  const int Ks[4] = {512, 256, 256, 256};
  const int Ns[4] = {256, 256, 256, 16};
  for (int l = 0; l < 4; l++) {
    if (l > 0) {
      float coeff = (l == 1) ? 1.0f
                  : (l == 2) ? logf(1.0f / 27.0f + 1.0f)
                             : logf(1.0f / 64.0f + 1.0f);
      axpy_elu<<<(NNODE * 256) / 256, 256, 0, stream>>>(inner, prev, coeff, NNODE * 256);
      cvt_bf16<<<(NNODE * 256) / 256, 256, 0, stream>>>(prev, hbf, NNODE * 256);
    }
    const __bf16* Ain = (l == 0) ? xbf : hbf;
    int K = Ks[l], Nc = Ns[l];
    transpose_f32_bf16<<<(K * Nc + 255) / 256, 256, 0, stream>>>(W[l], wt, K, Nc);
    gemm_bias<<<((NNODE / 16) * (Nc / 16)) / 8, 256, 0, stream>>>(Ain, wt, B[l], whf,
                                                                  NNODE, Nc, K);
    rowpack<<<NNODE, 256, 0, stream>>>(whf, Nc, whbf, sqn);
    transpose_f32_bf16<<<(NNODE * Nc + 255) / 256, 256, 0, stream>>>(whf, whtbf,
                                                                     NNODE, Nc);
    if (Nc == 256)
      fused_att_prop<256><<<NNODE / 16 / 8, 256, 0, stream>>>(
          adj, dg, sqn, whbf, whtbf, whf, inner, dc + 2 * l, cf + 3 * l, sp + l);
    else
      fused_att_prop<16><<<NNODE / 16 / 8, 256, 0, stream>>>(
          adj, dg, sqn, whbf, whtbf, whf, inner, dc + 2 * l, cf + 3 * l, sp + l);
  }
  logsoftmax16<<<NNODE / 256, 256, 0, stream>>>(inner, out);
}
